// Seq2Seq_with_attention_56719338111704
// MI455X (gfx1250) — matrix operand order repaired and checked by reading, not compiled, not run on hardware
//
#include <hip/hip_runtime.h>
#include <stdint.h>

// Seq2Seq-with-attention, MI455X (gfx1250, wave32, WMMA + async-to-LDS).
// - bf16-ize Uk and x so both co-reside in the 192MB L2 (67MB + 67MB): after
//   step 1 the attention stream is pure L2 traffic, never HBM.
// - persistent blocks: 16 batch rows/block, all 64 steps, state in LDS.
// - all shared-weight matmuls -> v_wmma_f32_16x16x32_bf16.
// - attention Uk stream double-buffered in LDS via global_load_async_to_lds_b128
//   (ASYNCcnt), overlapping DMA of row v+1 with tanh/FMA work on row v.

#define B_    512
#define WIN_  512
#define DIN_  128
#define HID_  256
#define DOUT_ 64
#define PLEN_ 64
#define ROWS_ 16

typedef __bf16 bf16;
typedef __attribute__((ext_vector_type(16))) __bf16 v16bf;
typedef __attribute__((ext_vector_type(8)))  float  v8f;
typedef __attribute__((ext_vector_type(8)))  __bf16 bf16x8;
typedef __attribute__((ext_vector_type(2)))  __bf16 bf16x2;

__device__ __forceinline__ float fast_rcp(float x) { return __builtin_amdgcn_rcpf(x); }
__device__ __forceinline__ float fast_exp(float x) { return __expf(x); }
// tanh(x) = 1 - 2/(1+exp(2x)); saturates correctly for |x| large.
__device__ __forceinline__ float fast_tanh(float x) { return 1.f - 2.f * fast_rcp(1.f + fast_exp(2.f * x)); }
__device__ __forceinline__ float fast_sig(float x)  { return fast_rcp(1.f + fast_exp(-x)); }

__device__ __forceinline__ v8f wmma_bf16(v16bf a, v16bf b, v8f c) {
  // 8 args: (neg_a, A, neg_b, B, c_mod, C, reuse_a, reuse_b)
  return __builtin_amdgcn_wmma_f32_16x16x32_bf16(false, a, false, b, (short)0, c, false, false);
}

// A-matrix fragment (16x32, 16-bit): ISA 7.12.2 layout.
template <typename F>
__device__ __forceinline__ v16bf load_a_frag(F f) {
  const int l = threadIdx.x & 31;
  const int m = l & 15;
  const int kb = (l < 16) ? 0 : 8;
  v16bf a;
#pragma unroll
  for (int i = 0; i < 16; ++i) {
    const int k = kb + ((i < 8) ? i : (i + 8));
    a[i] = f(m, k);
  }
  return a;
}

// B-matrix fragment (32x16, 16-bit): lane l -> N=l&15;
// lanes 0..15 hold K=0..7,16..23, lanes 16..31 hold K=8..15,24..31.
template <typename F>
__device__ __forceinline__ v16bf load_b_frag(F f) {
  const int l = threadIdx.x & 31;
  const int n = l & 15;
  const int kb = (l < 16) ? 0 : 8;
  v16bf b;
#pragma unroll
  for (int i = 0; i < 16; ++i) b[i] = f(kb + ((i < 8) ? i : (i + 8)), n);
  return b;
}

// Async DMA of 32 contiguous bytes per lane (2 x B128) global -> LDS.
__device__ __forceinline__ void async_copy_32B(unsigned lds_off, const bf16* g) {
  asm volatile("global_load_async_to_lds_b128 %0, %1, off"
               :: "v"(lds_off), "v"(g) : "memory");
  asm volatile("global_load_async_to_lds_b128 %0, %1, off"
               :: "v"(lds_off + 16u), "v"(g + 8) : "memory");
}
__device__ __forceinline__ void wait_async_le2() {
  asm volatile("s_wait_asynccnt 0x2" ::: "memory");
}
__device__ __forceinline__ void wait_async_0() {
  asm volatile("s_wait_asynccnt 0x0" ::: "memory");
}

// ---------------------------------------------------------------------------
__global__ void cvt_bf16_kernel(const float* __restrict__ src, bf16* __restrict__ dst, int n) {
  int i = blockIdx.x * blockDim.x + threadIdx.x;
  const int stride = gridDim.x * blockDim.x;
  for (; i < n; i += stride) dst[i] = (bf16)src[i];
}

// ---------------------------------------------------------------------------
// Uk[b,v,w] = sum_d U[v,d] * x[b,w,d]  (per-batch GEMM M=128,N=512,K=128), bf16 out.
__global__ __launch_bounds__(512) void uk_kernel(const float* __restrict__ x,
                                                 const bf16* __restrict__ Ub,
                                                 bf16* __restrict__ Ukb) {
  __shared__ bf16 Us[DIN_ * DIN_];  // 32 KB: U staged once per block
  const int b = blockIdx.x;
  for (int i = threadIdx.x; i < DIN_ * DIN_; i += blockDim.x) Us[i] = Ub[i];
  __syncthreads();

  const int wave = threadIdx.x >> 5;
  const int lane = threadIdx.x & 31;
  const float* xrow = x + (size_t)b * WIN_ * DIN_;

  for (int j = 0; j < 16; ++j) {         // 256 tiles / 16 waves
    const int tile = wave + 16 * j;
    const int mt = tile >> 5;            // 8 m-tiles (v)
    const int nt = tile & 31;            // 32 n-tiles (w)
    v8f acc = {};
    for (int ks = 0; ks < 4; ++ks) {     // K=128
      v16bf a = load_a_frag([&](int m, int k) -> bf16 {
        return Us[(mt * 16 + m) * DIN_ + ks * 32 + k];
      });
      v16bf bb = load_b_frag([&](int k, int n) -> bf16 {
        return (bf16)xrow[(size_t)(nt * 16 + n) * DIN_ + ks * 32 + k];
      });
      acc = wmma_bf16(a, bb, acc);
    }
    const int n = lane & 15;
    const int mb = (lane < 16) ? 0 : 8;
#pragma unroll
    for (int r = 0; r < 8; ++r) {
      const int m = mt * 16 + r + mb;
      Ukb[((size_t)b * DIN_ + m) * WIN_ + nt * 16 + n] = (bf16)acc[r];
    }
  }
}

// ---------------------------------------------------------------------------
// Persistent recurrent kernel: one block = 16 batch rows, 512 threads (16 waves),
// all 64 time steps. State + async staging buffers live in LDS (~117 KB).
__global__ __launch_bounds__(512) void seq_kernel(
    const bf16* __restrict__ Ukb, const bf16* __restrict__ xb,
    const bf16* __restrict__ Wb, const bf16* __restrict__ Wihb,
    const bf16* __restrict__ Whhb, const bf16* __restrict__ Wdb,
    const float* __restrict__ V, const float* __restrict__ b_ih,
    const float* __restrict__ b_hh, const float* __restrict__ bd,
    float* __restrict__ out) {
  __shared__ float s[ROWS_][HID_];         // 16 KB
  __shared__ float c[ROWS_][HID_];         // 16 KB
  __shared__ float y[ROWS_][DOUT_];        // 4 KB
  __shared__ float ctx[ROWS_][DIN_];       // 8 KB
  __shared__ float Wq[ROWS_][DIN_];        // 8 KB
  __shared__ float ew[ROWS_][WIN_];        // 32 KB (softmax weights)
  __shared__ bf16  ukbuf[ROWS_][2][WIN_];  // 32 KB async double buffer (per-wave)
  __shared__ float vv[DIN_];

  const int tid = threadIdx.x;
  const int wave = tid >> 5;
  const int lane = tid & 31;
  const int r0 = blockIdx.x * ROWS_;

  for (int i = tid; i < ROWS_ * HID_; i += 512) { (&s[0][0])[i] = 0.f; (&c[0][0])[i] = 0.f; }
  for (int i = tid; i < ROWS_ * DOUT_; i += 512) (&y[0][0])[i] = 0.f;
  if (tid < DIN_) vv[tid] = V[tid];
  __syncthreads();

  float* outY = out;
  float* outW = out + (size_t)B_ * PLEN_ * DOUT_;

  for (int t = 0; t < PLEN_; ++t) {
    // ---- Wq = [s|c] @ W^T : M=16, N=128, K=512 (waves 0..7, one n-tile each)
    if (wave < 8) {
      const int n0 = wave * 16;
      v8f acc = {};
      for (int ks = 0; ks < 16; ++ks) {
        v16bf a = load_a_frag([&](int m, int k) -> bf16 {
          const int kk = ks * 32 + k;
          return (bf16)((kk < HID_) ? s[m][kk] : c[m][kk - HID_]);
        });
        v16bf bb = load_b_frag([&](int k, int n) -> bf16 {
          return Wb[(size_t)(n0 + n) * (2 * HID_) + ks * 32 + k];
        });
        acc = wmma_bf16(a, bb, acc);
      }
      const int n = lane & 15, mb = (lane < 16) ? 0 : 8;
#pragma unroll
      for (int r = 0; r < 8; ++r) Wq[r + mb][n0 + n] = acc[r];
    }
    __syncthreads();

    // ---- attention scores + softmax: one wave per batch row. Each lane owns
    // 16 contiguous w values. Uk row v+1 streams into LDS via the async DMA
    // engine while row v is consumed (buffer is wave-private: no barriers,
    // only s_wait_asynccnt).
    {
      const int m = wave;
      const bf16* ukrow = Ukb + (size_t)(r0 + m) * DIN_ * WIN_;
      const int w0 = lane * 16;
      bf16* buf0 = &ukbuf[m][0][w0];
      bf16* buf1 = &ukbuf[m][1][w0];
      const unsigned lo0 = (unsigned)(uintptr_t)buf0;
      const unsigned lo1 = (unsigned)(uintptr_t)buf1;

      async_copy_32B(lo0, ukrow + w0);  // prime row v=0

      float e[16];
#pragma unroll
      for (int j = 0; j < 16; ++j) e[j] = 0.f;

      for (int v = 0; v < DIN_; ++v) {
        if (v + 1 < DIN_) {
          async_copy_32B(((v + 1) & 1) ? lo1 : lo0,
                         ukrow + (size_t)(v + 1) * WIN_ + w0);
          wait_async_le2();  // the two in-flight are row v+1; row v is done
        } else {
          wait_async_0();
        }
        const bf16x8* p = (const bf16x8*)((v & 1) ? buf1 : buf0);
        bf16x8 u0 = p[0], u1 = p[1];
        const float wqv = Wq[m][v];
        const float vw = vv[v];
#pragma unroll
        for (int j = 0; j < 8; ++j) {
          e[j]     += vw * fast_tanh(wqv + (float)u0[j]);
          e[8 + j] += vw * fast_tanh(wqv + (float)u1[j]);
        }
      }
      float mx = e[0];
#pragma unroll
      for (int j = 1; j < 16; ++j) mx = fmaxf(mx, e[j]);
      for (int off = 16; off >= 1; off >>= 1) mx = fmaxf(mx, __shfl_xor(mx, off, 32));
      float sum = 0.f;
#pragma unroll
      for (int j = 0; j < 16; ++j) { e[j] = fast_exp(e[j] - mx); sum += e[j]; }
      for (int off = 16; off >= 1; off >>= 1) sum += __shfl_xor(sum, off, 32);
      const float inv = fast_rcp(sum);
      float* ow = outW + ((size_t)(r0 + m) * PLEN_ + t) * WIN_ + w0;
#pragma unroll
      for (int j = 0; j < 16; ++j) {
        const float wt = e[j] * inv;
        ew[m][w0 + j] = wt;
        ow[j] = wt;
      }
    }
    __syncthreads();

    // ---- ctx = w @ x[b] : one wave per row, 4 d-values per lane (bf16x2 loads).
    {
      const int m = wave;
      const bf16* xr = xb + (size_t)(r0 + m) * WIN_ * DIN_;
      const int d0 = lane * 2, d1 = (lane + 32) * 2;
      float a0 = 0.f, a1 = 0.f, a2 = 0.f, a3 = 0.f;
      for (int w = 0; w < WIN_; ++w) {
        const float wt = ew[m][w];
        const bf16* px = xr + (size_t)w * DIN_;
        bf16x2 p0 = *(const bf16x2*)(px + d0);
        bf16x2 p1 = *(const bf16x2*)(px + d1);
        a0 += wt * (float)p0[0];
        a1 += wt * (float)p0[1];
        a2 += wt * (float)p1[0];
        a3 += wt * (float)p1[1];
      }
      ctx[m][d0] = a0; ctx[m][d0 + 1] = a1; ctx[m][d1] = a2; ctx[m][d1 + 1] = a3;
    }
    __syncthreads();

    // ---- gates = [y|ctx]@W_ih^T + s@W_hh^T (+biases), LSTM pointwise in regs.
    // Wave w owns n-tiles {w, w+16, w+32, w+48} == i/f/g/o for h in [16w,16w+16).
    {
      v8f gi = {}, gf = {}, gg = {}, go = {};
      for (int ks = 0; ks < 6; ++ks) {  // K=192 : inp=[y|ctx]
        v16bf a = load_a_frag([&](int m, int k) -> bf16 {
          const int kk = ks * 32 + k;
          return (bf16)((kk < DOUT_) ? y[m][kk] : ctx[m][kk - DOUT_]);
        });
#pragma unroll
        for (int g = 0; g < 4; ++g) {
          const int n0 = wave * 16 + g * HID_;
          v16bf bb = load_b_frag([&](int k, int n) -> bf16 {
            return Wihb[(size_t)(n0 + n) * (DIN_ + DOUT_) + ks * 32 + k];
          });
          v8f* accp = (g == 0) ? &gi : (g == 1) ? &gf : (g == 2) ? &gg : &go;
          *accp = wmma_bf16(a, bb, *accp);
        }
      }
      for (int ks = 0; ks < 8; ++ks) {  // K=256 : s
        v16bf a = load_a_frag([&](int m, int k) -> bf16 { return (bf16)s[m][ks * 32 + k]; });
#pragma unroll
        for (int g = 0; g < 4; ++g) {
          const int n0 = wave * 16 + g * HID_;
          v16bf bb = load_b_frag([&](int k, int n) -> bf16 {
            return Whhb[(size_t)(n0 + n) * HID_ + ks * 32 + k];
          });
          v8f* accp = (g == 0) ? &gi : (g == 1) ? &gf : (g == 2) ? &gg : &go;
          *accp = wmma_bf16(a, bb, *accp);
        }
      }
      const int h = wave * 16 + (lane & 15);
      const float bI = b_ih[h] + b_hh[h];
      const float bF = b_ih[HID_ + h] + b_hh[HID_ + h];
      const float bG = b_ih[2 * HID_ + h] + b_hh[2 * HID_ + h];
      const float bO = b_ih[3 * HID_ + h] + b_hh[3 * HID_ + h];
      const int mb = (lane < 16) ? 0 : 8;
      __syncthreads();  // all waves done reading old s before it is overwritten
#pragma unroll
      for (int r = 0; r < 8; ++r) {
        const int m = r + mb;
        const float ig = fast_sig(gi[r] + bI);
        const float fg = fast_sig(gf[r] + bF);
        const float gt = fast_tanh(gg[r] + bG);
        const float og = fast_sig(go[r] + bO);
        const float cn = fg * c[m][h] + ig * gt;
        const float sn = og * fast_tanh(cn);
        c[m][h] = cn;
        s[m][h] = sn;
      }
    }
    __syncthreads();

    // ---- y = s_new @ Wd^T + bd : M=16, N=64, K=256 (waves 0..3)
    if (wave < 4) {
      const int n0 = wave * 16;
      v8f acc = {};
      for (int ks = 0; ks < 8; ++ks) {
        v16bf a = load_a_frag([&](int m, int k) -> bf16 { return (bf16)s[m][ks * 32 + k]; });
        v16bf bb = load_b_frag([&](int k, int n) -> bf16 {
          return Wdb[(size_t)(n0 + n) * HID_ + ks * 32 + k];
        });
        acc = wmma_bf16(a, bb, acc);
      }
      const int n = lane & 15, mb = (lane < 16) ? 0 : 8;
#pragma unroll
      for (int r = 0; r < 8; ++r) {
        const int m = r + mb;
        const float yv = acc[r] + bd[n0 + n];
        y[m][n0 + n] = yv;
        outY[((size_t)(r0 + m) * PLEN_ + t) * DOUT_ + n0 + n] = yv;
      }
    }
    __syncthreads();
  }
}

// ---------------------------------------------------------------------------
extern "C" void kernel_launch(void* const* d_in, const int* in_sizes, int n_in,
                              void* d_out, int out_size, void* d_ws, size_t ws_size,
                              hipStream_t stream) {
  const float* x    = (const float*)d_in[0];
  // d_in[1] = plength (==64, compile-time constant PLEN_)
  const float* V    = (const float*)d_in[2];
  const float* W    = (const float*)d_in[3];
  const float* U    = (const float*)d_in[4];
  const float* W_ih = (const float*)d_in[5];
  const float* W_hh = (const float*)d_in[6];
  const float* b_ih = (const float*)d_in[7];
  const float* b_hh = (const float*)d_in[8];
  const float* Wd   = (const float*)d_in[9];
  const float* bd   = (const float*)d_in[10];
  float* out = (float*)d_out;

  // Workspace layout (bf16), ~129 MB total:
  bf16* Ukb  = (bf16*)d_ws;                                   // 512*128*512
  bf16* xbf  = Ukb + (size_t)B_ * DIN_ * WIN_;                // 512*512*128
  bf16* Ub   = xbf + (size_t)B_ * WIN_ * DIN_;                // 128*128
  bf16* Wb   = Ub + DIN_ * DIN_;                              // 128*512
  bf16* Wihb = Wb + DIN_ * 2 * HID_;                          // 1024*192
  bf16* Whhb = Wihb + 4 * HID_ * (DIN_ + DOUT_);              // 1024*256
  bf16* Wdb  = Whhb + 4 * HID_ * HID_;                        // 64*256

  const int CT = 256;
  cvt_bf16_kernel<<<2048, CT, 0, stream>>>(x, xbf, B_ * WIN_ * DIN_);
  cvt_bf16_kernel<<<64, CT, 0, stream>>>(U, Ub, DIN_ * DIN_);
  cvt_bf16_kernel<<<256, CT, 0, stream>>>(W, Wb, DIN_ * 2 * HID_);
  cvt_bf16_kernel<<<768, CT, 0, stream>>>(W_ih, Wihb, 4 * HID_ * (DIN_ + DOUT_));
  cvt_bf16_kernel<<<1024, CT, 0, stream>>>(W_hh, Whhb, 4 * HID_ * HID_);
  cvt_bf16_kernel<<<64, CT, 0, stream>>>(Wd, Wdb, DOUT_ * HID_);

  uk_kernel<<<B_, 512, 0, stream>>>(x, Ub, Ukb);

  seq_kernel<<<B_ / ROWS_, 512, 0, stream>>>(Ukb, xbf, Wb, Wihb, Whhb, Wdb,
                                             V, b_ih, b_hh, bd, out);
}
